// QuantLinear_39084202394119
// MI455X (gfx1250) — matrix operand order repaired and checked by reading, not compiled, not run on hardware
//
#include <hip/hip_runtime.h>

// -------- vector types --------
typedef __attribute__((ext_vector_type(16))) _Float16 h16;
typedef __attribute__((ext_vector_type(8)))  _Float16 h8;
typedef __attribute__((ext_vector_type(4)))  _Float16 h4;
typedef __attribute__((ext_vector_type(2)))  _Float16 h2;
typedef __attribute__((ext_vector_type(2)))  __fp16   fp16x2;
typedef __attribute__((ext_vector_type(8)))  float    f8;
typedef __attribute__((ext_vector_type(4)))  float    f4;

#define K_DIM   4096
#define N_DIM   11008
#define BM      128
#define BN      256
#define BK      128           // == quant GROUP_SIZE: one (scale, zero) per stage
#define LDSW    136           // BK + 8 halves pad -> 272B row = 17*16B, conflict-free b128
#define NSTAGES (K_DIM / BK)  // 32

static __device__ __forceinline__ h2 pack2(float a, float b) {
#if __has_builtin(__builtin_amdgcn_cvt_pkrtz)
    fp16x2 t = __builtin_amdgcn_cvt_pkrtz(a, b);   // v_cvt_pk_rtz_f16_f32
    return __builtin_bit_cast(h2, t);
#else
    h2 r; r[0] = (_Float16)a; r[1] = (_Float16)b; return r;
#endif
}

__global__ __launch_bounds__(256)
void qlinear_wmma_f16(const float* __restrict__ x,
                      const int*   __restrict__ qweight,
                      const int*   __restrict__ qzeros,
                      const float* __restrict__ scales,
                      const float* __restrict__ bias,
                      float* __restrict__ out)
{
    extern __shared__ char smem_raw[];
    _Float16* As = (_Float16*)smem_raw;     // [BM][LDSW]  x tile, f16, K-contiguous
    _Float16* Bs = As + BM * LDSW;          // [BN][LDSW]  dequant W tile, f16, K-contiguous

    const int tid  = threadIdx.x;
    const int lane = tid & 31;
    const int wave = tid >> 5;

    const int n0 = blockIdx.x * BN;
    const int m0 = blockIdx.y * BM;

    // dequant assignment: each thread owns one N column, all 16 packed-K rows
    const int nGlob = n0 + tid;

    // wave tiling: 2 waves along M (64 rows) x 4 waves along N (64 cols)
    const int mWave = (wave & 1) * 64;
    const int nWave = (wave >> 1) * 64;

    const int r16  = lane & 15;             // row/col within fragment
    const int half = lane >> 4;             // lane half selects K sub-range

    f8 acc[4][4];
#pragma unroll
    for (int i = 0; i < 4; ++i)
#pragma unroll
        for (int j = 0; j < 4; ++j)
#pragma unroll
            for (int e = 0; e < 8; ++e) acc[i][j][e] = 0.0f;

    for (int stage = 0; stage < NSTAGES; ++stage) {
        const int k0 = stage * BK;
        const int g  = stage;               // BK == GROUP_SIZE

        __syncthreads();                    // previous compute done reading LDS

        // ---- stage x tile: BM x BK f32 -> f16 into As ----
#pragma unroll
        for (int it = 0; it < 16; ++it) {
            const int idx = it * 256 + tid;         // float4 index in tile
            const int m   = idx >> 5;               // 32 float4 per row
            const int kq  = (idx & 31) << 2;
            const f4 v = *(const f4*)(x + (size_t)(m0 + m) * K_DIM + k0 + kq);
            const h2 p0 = pack2(v[0], v[1]);
            const h2 p1 = pack2(v[2], v[3]);
            h4 hv; hv[0] = p0[0]; hv[1] = p0[1]; hv[2] = p1[0]; hv[3] = p1[1];
            *(h4*)(As + m * LDSW + kq) = hv;        // ds_store_b64
        }

        // ---- dequant weight tile: 16 x BN int32 -> f16 [N][K] into Bs ----
        {
            const float s    = scales[(size_t)g * N_DIM + nGlob];
            const int   zraw = qzeros[(size_t)g * (N_DIM / 8) + (nGlob >> 3)];
            const float nsz  = -s * (float)((zraw >> ((nGlob & 7) * 4)) & 0xF);
            const int* qcol  = qweight + (size_t)(stage * (BK / 8)) * N_DIM + nGlob;
            _Float16*  brow  = Bs + (size_t)tid * LDSW;
#pragma unroll
            for (int j = 0; j < 16; ++j) {
                const unsigned int w  = (unsigned int)qcol[(size_t)j * N_DIM];
                const unsigned int we = w & 0x0F0F0F0Fu;          // nibbles 0,2,4,6
                const unsigned int wo = (w >> 4) & 0x0F0F0F0Fu;   // nibbles 1,3,5,7
                // (float)(byte) patterns -> v_cvt_f32_ubyte{0..3}; one v_fma_f32 each
                const float f0 = fmaf((float)(we & 0xFFu),         s, nsz);
                const float f1 = fmaf((float)(wo & 0xFFu),         s, nsz);
                const float f2 = fmaf((float)((we >> 8) & 0xFFu),  s, nsz);
                const float f3 = fmaf((float)((wo >> 8) & 0xFFu),  s, nsz);
                const float f4v = fmaf((float)((we >> 16) & 0xFFu), s, nsz);
                const float f5 = fmaf((float)((wo >> 16) & 0xFFu), s, nsz);
                const float f6 = fmaf((float)(we >> 24),           s, nsz);
                const float f7 = fmaf((float)(wo >> 24),           s, nsz);
                const h2 p0 = pack2(f0, f1), p1 = pack2(f2, f3);
                const h2 p2 = pack2(f4v, f5), p3 = pack2(f6, f7);
                h8 hv;
                hv[0] = p0[0]; hv[1] = p0[1]; hv[2] = p1[0]; hv[3] = p1[1];
                hv[4] = p2[0]; hv[5] = p2[1]; hv[6] = p3[0]; hv[7] = p3[1];
                *(h8*)(brow + j * 8) = hv;                        // ds_store_b128
            }
        }

        __syncthreads();

        // ---- prefetch next stage while WMMAs run (global_prefetch_b8) ----
        if (stage + 1 < NSTAGES) {
            const int* qn = qweight + (size_t)((stage + 1) * (BK / 8)) * N_DIM + nGlob;
            __builtin_prefetch(qn,             0, 1);
            __builtin_prefetch(qn + (size_t)4  * N_DIM, 0, 1);
            __builtin_prefetch(qn + (size_t)8  * N_DIM, 0, 1);
            __builtin_prefetch(qn + (size_t)12 * N_DIM, 0, 1);
            __builtin_prefetch(x + (size_t)(m0 + (tid & 127)) * K_DIM + (stage + 1) * BK + ((tid >> 7) * 64), 0, 1);
        }

        // ---- compute: 4 k-steps of 16x16x32; 4(M) x 4(N) frags per wave ----
#pragma unroll
        for (int kk = 0; kk < 4; ++kk) {
            const int kb = kk * 32;

            h16 afrag[4];
#pragma unroll
            for (int i = 0; i < 4; ++i) {
                // 16-bit A 16x32 layout: lanes 0-15 K[0..7]+K[16..23], lanes 16-31 K[8..15]+K[24..31]
                const _Float16* p = As + (mWave + i * 16 + r16) * LDSW + kb + half * 8;
                const h8 lo = *(const h8*)p;          // ds_load_b128
                const h8 hi = *(const h8*)(p + 16);   // ds_load_b128
#pragma unroll
                for (int e = 0; e < 8; ++e) { afrag[i][e] = lo[e]; afrag[i][8 + e] = hi[e]; }
            }

#pragma unroll
            for (int jn = 0; jn < 4; ++jn) {
                // 16-bit B 32x16 layout: lanes 0-15 K[0..7]+K[16..23], lanes 16-31 K[8..15]+K[24..31], N = lane&15
                const _Float16* p = Bs + (nWave + jn * 16 + r16) * LDSW + kb + half * 8;
                const h8 lo = *(const h8*)p;
                const h8 hi = *(const h8*)(p + 16);
                h16 bfrag;
#pragma unroll
                for (int e = 0; e < 8; ++e) { bfrag[e] = lo[e]; bfrag[8 + e] = hi[e]; }

#pragma unroll
                for (int i = 0; i < 4; ++i) {
                    acc[i][jn] = __builtin_amdgcn_wmma_f32_16x16x32_f16(
                        /*neg_a=*/false, afrag[i],
                        /*neg_b=*/false, bfrag,
                        /*c_mod=*/(short)0, acc[i][jn],
                        /*reuse_a=*/false, /*reuse_b=*/false);
                }
            }
        }
    }

    // ---- epilogue: C/D layout -> global, add bias ----
    // lane 0-15: M rows base+0..7 ; lane 16-31: M rows base+8..15 ; N = lane&15
#pragma unroll
    for (int jn = 0; jn < 4; ++jn) {
        const int n   = n0 + nWave + jn * 16 + r16;
        const float bv = bias[n];
#pragma unroll
        for (int i = 0; i < 4; ++i) {
            const int mbase = m0 + mWave + i * 16 + half * 8;
#pragma unroll
            for (int r = 0; r < 8; ++r) {
                out[(size_t)(mbase + r) * N_DIM + n] = acc[i][jn][r] + bv;
            }
        }
    }
}

extern "C" void kernel_launch(void* const* d_in, const int* in_sizes, int n_in,
                              void* d_out, int out_size, void* d_ws, size_t ws_size,
                              hipStream_t stream) {
    const float* xp   = (const float*)d_in[0];
    const int*   qw   = (const int*)d_in[1];
    const int*   qz   = (const int*)d_in[2];
    const float* sc   = (const float*)d_in[3];
    const float* bi   = (const float*)d_in[4];
    float*       outp = (float*)d_out;

    const int M = in_sizes[0] / K_DIM;       // 4*512 = 2048
    dim3 grid(N_DIM / BN, M / BM);           // 43 x 16
    const size_t shmem = (size_t)(BM + BN) * LDSW * sizeof(_Float16); // ~102 KB dynamic LDS

    // opt in to >64KB dynamic LDS (WGP has 320KB); idempotent, capture-safe
    (void)hipFuncSetAttribute((const void*)qlinear_wmma_f16,
                              hipFuncAttributeMaxDynamicSharedMemorySize,
                              (int)shmem);

    qlinear_wmma_f16<<<grid, 256, shmem, stream>>>(xp, qw, qz, sc, bi, outp);
}
